// AgentAttention_77197742178864
// MI455X (gfx1250) — compile-verified, matrix operand order checked by reading, not run on hardware
//
#include <hip/hip_runtime.h>
#include <hip/hip_bf16.h>
#include <stdint.h>

typedef __bf16 bf16_t;
typedef __attribute__((ext_vector_type(16))) __bf16 v16bf;
typedef __attribute__((ext_vector_type(8)))  float  v8f;
typedef uint32_t u32x4 __attribute__((ext_vector_type(4)));
typedef uint32_t u32x8 __attribute__((ext_vector_type(8)));

#define B_   32
#define N_   3136
#define C_   512
#define HH   8
#define HD   64
#define AG   49
#define AGP  64
#define HW   56
#define KT   32

// ---------- helpers -------------------------------------------------------

__device__ __forceinline__ v8f zero8() {
  v8f z;
#pragma unroll
  for (int i = 0; i < 8; ++i) z[i] = 0.f;
  return z;
}

// Fragment load from a tile stored row-major [row][k] (value = p[row*ld + k]).
// ISA 16-bit layout: lane<16 holds row=lane, element e -> K = e + (e>=8?8:0);
// lane>=16 holds same row with K offset +8.
__device__ __forceinline__ v16bf frag_rowk(const bf16_t* p, int ld) {
  int lane = threadIdx.x & 31;
  int row  = lane & 15;
  int kb   = (lane >> 4) << 3;
  v16bf f;
#pragma unroll
  for (int e = 0; e < 16; ++e) {
    int k = e + ((e >= 8) ? 8 : 0) + kb;
    f[e] = p[row * ld + k];
  }
  return f;
}

// B fragment from a tile stored K-major [k][n] (value = p[k*ld + n])
__device__ __forceinline__ v16bf frag_kn(const bf16_t* p, int ld) {
  int lane = threadIdx.x & 31;
  int n    = lane & 15;
  int kb   = (lane >> 4) << 3;
  v16bf f;
#pragma unroll
  for (int e = 0; e < 16; ++e) {
    int k = e + ((e >= 8) ? 8 : 0) + kb;
    f[e] = p[k * ld + n];
  }
  return f;
}

__device__ __forceinline__ v8f wmma_bf(v16bf a, v16bf b, v8f c) {
  return __builtin_amdgcn_wmma_f32_16x16x32_bf16(false, a, false, b, (short)0, c,
                                                 false, false);
}

__device__ __forceinline__ float half_max16(float v) {
#pragma unroll
  for (int m = 1; m <= 8; m <<= 1) v = fmaxf(v, __shfl_xor(v, m, 32));
  return v;
}
__device__ __forceinline__ float half_sum16(float v) {
#pragma unroll
  for (int m = 1; m <= 8; m <<= 1) v += __shfl_xor(v, m, 32);
  return v;
}

// TDM: DMA a 2D tile (rows x tile_dw dwords, row stride stride_dw dwords)
// from global memory into LDS. D# layout per CDNA5 ISA 8.3/8.4.
__device__ __forceinline__ void tdm_load_2d(unsigned lds_addr, const void* gptr,
                                            unsigned tile_dw, unsigned rows,
                                            unsigned stride_dw, unsigned tdim0,
                                            unsigned trows) {
  uint64_t ga = (uint64_t)(uintptr_t)gptr;
  u32x4 g0;
  g0[0] = 1u;                                        // count=1, user mode
  g0[1] = lds_addr;                                  // lds_addr
  g0[2] = (uint32_t)ga;                              // global_addr[31:0]
  g0[3] = ((uint32_t)(ga >> 32) & 0x01FFFFFFu) | 0x80000000u;  // addr hi | type=2
  u32x8 g1;
  g1[0] = 2u << 16;                                  // data_size = 4B
  g1[1] = (tdim0 & 0xFFFFu) << 16;                   // tensor_dim0[15:0]
  g1[2] = (tdim0 >> 16) | ((trows & 0xFFFFu) << 16); // tensor_dim0 hi | dim1 lo
  g1[3] = (trows >> 16) | (tile_dw << 16);           // tensor_dim1 hi | tile_dim0
  g1[4] = rows;                                      // tile_dim1 (tile_dim2=0)
  g1[5] = stride_dw;                                 // tensor_dim0_stride[31:0]
  g1[6] = 0;
  g1[7] = 0;
  asm volatile("tensor_load_to_lds %0, %1" :: "s"(g0), "s"(g1) : "memory");
}

// async global -> LDS (128-bit per lane), tracked with ASYNCcnt
__device__ __forceinline__ void async_ld_b128(unsigned lds_off, const void* g) {
  asm volatile("global_load_async_to_lds_b128 %0, %1, off"
               :: "v"(lds_off), "v"((uint64_t)(uintptr_t)g) : "memory");
}
__device__ __forceinline__ unsigned lds_off32(const void* p) {
  return (unsigned)(uintptr_t)p;
}

// ---------- kernels -------------------------------------------------------

__global__ void cvt_bf16(const float* __restrict__ in, bf16_t* __restrict__ out, int n) {
  int i = blockIdx.x * blockDim.x + threadIdx.x;
  if (i < n) out[i] = (bf16_t)in[i];
}

// C[M,1536] = Xb[M,512] * Wb[1536,512]^T ; split columns into q/k/v (bf16)
// TDM double-buffered staging.
__global__ __launch_bounds__(128) void gemm_qkv(const bf16_t* __restrict__ x,
                                                const bf16_t* __restrict__ w,
                                                bf16_t* __restrict__ q,
                                                bf16_t* __restrict__ k,
                                                bf16_t* __restrict__ v) {
  __shared__ bf16_t Al[2][64 * KT];
  __shared__ bf16_t Bl[2][64 * KT];
  int mt = blockIdx.x, nt = blockIdx.y;
  int t = threadIdx.x, wv = t >> 5, lane = t & 31;
  v8f acc[4];
#pragma unroll
  for (int j = 0; j < 4; ++j) acc[j] = zero8();

  const bf16_t* ga0 = x + (size_t)(mt * 64) * C_;
  const bf16_t* gb0 = w + (size_t)(nt * 64) * C_;
  const int niter = C_ / KT;  // 16

  if (wv == 0) {
    tdm_load_2d(lds_off32(&Al[0][0]), ga0, KT / 2, 64, C_ / 2, C_ / 2, 1u << 20);
    tdm_load_2d(lds_off32(&Bl[0][0]), gb0, KT / 2, 64, C_ / 2, C_ / 2, 1u << 20);
  }
  for (int it = 0; it < niter; ++it) {
    int cur = it & 1;
    __syncthreads();  // everyone done with buffer cur^1
    if (wv == 0) {
      if (it + 1 < niter) {
        tdm_load_2d(lds_off32(&Al[cur ^ 1][0]), ga0 + (it + 1) * KT, KT / 2, 64,
                    C_ / 2, C_ / 2, 1u << 20);
        tdm_load_2d(lds_off32(&Bl[cur ^ 1][0]), gb0 + (it + 1) * KT, KT / 2, 64,
                    C_ / 2, C_ / 2, 1u << 20);
        __builtin_amdgcn_s_wait_tensorcnt(2);  // chunk `it` complete
      } else {
        __builtin_amdgcn_s_wait_tensorcnt(0);
      }
    }
    __syncthreads();
    v16bf a = frag_rowk(&Al[cur][(wv * 16) * KT], KT);
#pragma unroll
    for (int j = 0; j < 4; ++j)
      acc[j] = wmma_bf(a, frag_rowk(&Bl[cur][(j * 16) * KT], KT), acc[j]);
  }

  int rbase = mt * 64 + wv * 16 + ((lane >> 4) << 3);
#pragma unroll
  for (int j = 0; j < 4; ++j) {
    int col = nt * 64 + j * 16 + (lane & 15);
    int g = col >> 9, cc = col & 511;
    bf16_t* dst = (g == 0) ? q : (g == 1) ? k : v;
#pragma unroll
    for (int r = 0; r < 8; ++r)
      dst[(size_t)(rbase + r) * C_ + cc] = (bf16_t)acc[j][r];
  }
}

// agent[b, a, c] = SCALE * mean_{8x8 block}(q) ; padded to AGP rows (zeros)
__global__ void agent_pool(const bf16_t* __restrict__ q, bf16_t* __restrict__ agent) {
  int i = blockIdx.x * blockDim.x + threadIdx.x;
  if (i >= B_ * AGP * C_) return;
  int c = i & (C_ - 1);
  int a = (i >> 9) & 63;
  int b = i >> 15;
  float s = 0.f;
  if (a < AG) {
    int p1 = a / 7, p2 = a % 7;
#pragma unroll 2
    for (int dy = 0; dy < 8; ++dy)
      for (int dx = 0; dx < 8; ++dx) {
        int n = (p1 * 8 + dy) * HW + (p2 * 8 + dx);
        s += (float)q[((size_t)b * N_ + n) * C_ + c];
      }
    s *= (1.f / 64.f) * 0.125f;  // mean * SCALE (SCALE = 64^-0.5)
  }
  agent[i] = (bf16_t)s;
}

// Fused: agent_attn = softmax(agent @ K^T) ; agent_v = agent_attn @ V
// one block per (b,h); 4 waves x 16 agent rows; flash-style over 3136 keys.
// K/V chunks staged with async global->LDS loads, double-buffered.
__global__ __launch_bounds__(128) void agent_attn(const bf16_t* __restrict__ agent,
                                                  const bf16_t* __restrict__ kmat,
                                                  const bf16_t* __restrict__ vmat,
                                                  bf16_t* __restrict__ agentv) {
  __shared__ bf16_t Kl[2][32 * HD];
  __shared__ bf16_t Vl[2][32 * HD];
  __shared__ bf16_t Pl[4][16 * 32];
  int b = blockIdx.x >> 3, h = blockIdx.x & 7;
  int t = threadIdx.x, wv = t >> 5, lane = t & 31;

  const bf16_t* ag = agent + ((size_t)b * AGP + wv * 16) * C_ + h * HD;
  v16bf a0 = frag_rowk(ag, C_);       // agent rows, d 0..31
  v16bf a1 = frag_rowk(ag + 32, C_);  // agent rows, d 32..63

  float rm[8], rl[8];
  v8f acc[4];
#pragma unroll
  for (int r = 0; r < 8; ++r) { rm[r] = -1e30f; rl[r] = 0.f; }
#pragma unroll
  for (int j = 0; j < 4; ++j) acc[j] = zero8();

  int srow = t >> 2, sdof = (t & 3) * 16;
  const bf16_t* gk0 = kmat + ((size_t)b * N_ + srow) * C_ + h * HD + sdof;
  const bf16_t* gv0 = vmat + ((size_t)b * N_ + srow) * C_ + h * HD + sdof;
  const int nchunks = N_ / 32;  // 98

  {  // prologue: chunk 0 into buffer 0 (4 async b128 per thread)
    unsigned lk = lds_off32(&Kl[0][srow * HD + sdof]);
    unsigned lv = lds_off32(&Vl[0][srow * HD + sdof]);
    async_ld_b128(lk, gk0);       async_ld_b128(lk + 16, gk0 + 8);
    async_ld_b128(lv, gv0);       async_ld_b128(lv + 16, gv0 + 8);
  }

  for (int it = 0; it < nchunks; ++it) {
    int cur = it & 1;
    __syncthreads();  // everyone done reading buffer cur^1
    if (it + 1 < nchunks) {
      const bf16_t* gk = gk0 + (size_t)(it + 1) * 32 * C_;
      const bf16_t* gv = gv0 + (size_t)(it + 1) * 32 * C_;
      unsigned lk = lds_off32(&Kl[cur ^ 1][srow * HD + sdof]);
      unsigned lv = lds_off32(&Vl[cur ^ 1][srow * HD + sdof]);
      async_ld_b128(lk, gk);      async_ld_b128(lk + 16, gk + 8);
      async_ld_b128(lv, gv);      async_ld_b128(lv + 16, gv + 8);
      asm volatile("s_wait_asynccnt 0x4" ::: "memory");  // chunk `it` done
    } else {
      asm volatile("s_wait_asynccnt 0x0" ::: "memory");
    }
    __syncthreads();

    // scores: S0 = keys 0..15, S1 = keys 16..31 (accumulate over d = 2x32)
    const bf16_t* Kc = &Kl[cur][0];
    const bf16_t* Vc = &Vl[cur][0];
    v8f s0 = wmma_bf(a1, frag_rowk(Kc + 32, HD),
                     wmma_bf(a0, frag_rowk(Kc, HD), zero8()));
    v8f s1 = wmma_bf(a1, frag_rowk(Kc + 16 * HD + 32, HD),
                     wmma_bf(a0, frag_rowk(Kc + 16 * HD, HD), zero8()));

    // online softmax (C-layout: each (lane-half, element) pair is one row)
#pragma unroll
    for (int r = 0; r < 8; ++r) {
      float mx = half_max16(fmaxf(s0[r], s1[r]));
      float nm = fmaxf(rm[r], mx);
      float sc = __expf(rm[r] - nm);
      rm[r] = nm;
      float p0 = __expf(s0[r] - nm);
      float p1 = __expf(s1[r] - nm);
      rl[r] = rl[r] * sc + half_sum16(p0 + p1);
      s0[r] = p0;
      s1[r] = p1;
#pragma unroll
      for (int j = 0; j < 4; ++j) acc[j][r] *= sc;
    }

    // repack P (C-layout -> A-layout) through per-wave LDS slice
    {
      bf16_t* pl = &Pl[wv][0];
      int prow = (lane >> 4) << 3;
      int pcol = lane & 15;
#pragma unroll
      for (int r = 0; r < 8; ++r) {
        pl[(prow + r) * 32 + pcol]      = (bf16_t)s0[r];
        pl[(prow + r) * 32 + pcol + 16] = (bf16_t)s1[r];
      }
    }
    asm volatile("s_wait_dscnt 0" ::: "memory");
    v16bf pa = frag_rowk(&Pl[wv][0], 32);
#pragma unroll
    for (int j = 0; j < 4; ++j)
      acc[j] = wmma_bf(pa, frag_kn(Vc + j * 16, HD), acc[j]);
  }

  // finalize: normalize, write agent_v (zero-pad rows >= 49)
  int rbase = wv * 16 + ((lane >> 4) << 3);
#pragma unroll
  for (int j = 0; j < 4; ++j) {
    int d = j * 16 + (lane & 15);
#pragma unroll
    for (int r = 0; r < 8; ++r) {
      int row = rbase + r;
      float val = (row < AG) ? acc[j][r] / rl[r] : 0.f;
      agentv[(((size_t)b * HH + h) * AGP + row) * HD + d] = (bf16_t)val;
    }
  }
}

// depthwise 3x3 conv on v (bf16 in, bf16 out), bias included
__global__ void dwc_conv(const bf16_t* __restrict__ v, const float* __restrict__ w,
                         const float* __restrict__ bias, bf16_t* __restrict__ out) {
  int i = blockIdx.x * blockDim.x + threadIdx.x;
  if (i >= B_ * N_ * C_) return;
  int c = i & 511;
  int n = (i >> 9) % N_;
  int b = (i >> 9) / N_;
  int y = n / HW, x = n % HW;
  float s = bias[c];
#pragma unroll
  for (int dy = -1; dy <= 1; ++dy)
#pragma unroll
    for (int dx = -1; dx <= 1; ++dx) {
      int yy = y + dy, xx = x + dx;
      if (yy >= 0 && yy < HW && xx >= 0 && xx < HW)
        s += w[c * 9 + (dy + 1) * 3 + (dx + 1)] *
             (float)v[((size_t)b * N_ + yy * HW + xx) * C_ + c];
    }
  out[i] = (bf16_t)s;
}

// Fused: q_attn = softmax(q @ agent^T) ; out = q_attn @ agent_v + dwc -> bf16
__global__ __launch_bounds__(128) void q_attn(const bf16_t* __restrict__ qmat,
                                              const bf16_t* __restrict__ agent,
                                              const bf16_t* __restrict__ agentv,
                                              const bf16_t* __restrict__ dwc,
                                              bf16_t* __restrict__ outb) {
  __shared__ bf16_t Agl[AGP * HD];
  __shared__ bf16_t AVl[AGP * HD];
  __shared__ bf16_t Pl[4][16 * 64];
  int b = blockIdx.z, h = blockIdx.y;
  int t = threadIdx.x, wv = t >> 5, lane = t & 31;

  {  // stage agent + agent_v head slices (64x64 each) with async loads
    int row = t >> 1;
    int dof = (t & 1) * 32;
    const bf16_t* ga = agent + ((size_t)b * AGP + row) * C_ + h * HD + dof;
    const bf16_t* gv = agentv + (((size_t)b * HH + h) * AGP + row) * HD + dof;
    unsigned la = lds_off32(&Agl[row * HD + dof]);
    unsigned lv = lds_off32(&AVl[row * HD + dof]);
    async_ld_b128(la, ga);           async_ld_b128(la + 16, ga + 8);
    async_ld_b128(la + 32, ga + 16); async_ld_b128(la + 48, ga + 24);
    async_ld_b128(lv, gv);           async_ld_b128(lv + 16, gv + 8);
    async_ld_b128(lv + 32, gv + 16); async_ld_b128(lv + 48, gv + 24);
    asm volatile("s_wait_asynccnt 0x0" ::: "memory");
  }
  __syncthreads();

  int qt = blockIdx.x * 4 + wv;  // query tile (16 rows), 0..195
  const bf16_t* qp = qmat + ((size_t)b * N_ + qt * 16) * C_ + h * HD;
  v16bf a0 = frag_rowk(qp, C_);
  v16bf a1 = frag_rowk(qp + 32, C_);

  v8f s[4];
#pragma unroll
  for (int j = 0; j < 4; ++j)
    s[j] = wmma_bf(a1, frag_rowk(&Agl[j * 16 * HD + 32], HD),
                   wmma_bf(a0, frag_rowk(&Agl[j * 16 * HD], HD), zero8()));

  // mask padded agents, softmax over 49
#pragma unroll
  for (int j = 0; j < 4; ++j) {
    bool valid = (j * 16 + (lane & 15)) < AG;
#pragma unroll
    for (int r = 0; r < 8; ++r)
      if (!valid) s[j][r] = -1e30f;
  }
#pragma unroll
  for (int r = 0; r < 8; ++r) {
    float mx = half_max16(fmaxf(fmaxf(s[0][r], s[1][r]), fmaxf(s[2][r], s[3][r])));
    float sum = 0.f;
#pragma unroll
    for (int j = 0; j < 4; ++j) { s[j][r] = __expf(s[j][r] - mx); sum += s[j][r]; }
    float inv = 1.f / half_sum16(sum);
#pragma unroll
    for (int j = 0; j < 4; ++j) s[j][r] *= inv;
  }

  // repack P (16x64) through per-wave LDS slice
  {
    bf16_t* pl = &Pl[wv][0];
    int prow = (lane >> 4) << 3;
    int pcol = lane & 15;
#pragma unroll
    for (int j = 0; j < 4; ++j)
#pragma unroll
      for (int r = 0; r < 8; ++r)
        pl[(prow + r) * 64 + j * 16 + pcol] = (bf16_t)s[j][r];
  }
  asm volatile("s_wait_dscnt 0" ::: "memory");
  v16bf p0 = frag_rowk(&Pl[wv][0], 64);
  v16bf p1 = frag_rowk(&Pl[wv][32], 64);
  v8f acc[4];
#pragma unroll
  for (int j = 0; j < 4; ++j)
    acc[j] = wmma_bf(p1, frag_kn(&AVl[32 * HD + j * 16], HD),
                     wmma_bf(p0, frag_kn(&AVl[j * 16], HD), zero8()));

  // add dwc and write bf16
  int rbase = (lane >> 4) << 3;
#pragma unroll
  for (int j = 0; j < 4; ++j) {
    int c = h * HD + j * 16 + (lane & 15);
#pragma unroll
    for (int r = 0; r < 8; ++r) {
      size_t n = (size_t)b * N_ + qt * 16 + rbase + r;
      float val = acc[j][r] + (float)dwc[n * C_ + c];
      outb[n * C_ + c] = (bf16_t)val;
    }
  }
}

// final = outb[M,512] * Wproj[512,512]^T + bias -> fp32 (TDM staging)
__global__ __launch_bounds__(128) void gemm_proj(const bf16_t* __restrict__ x,
                                                 const bf16_t* __restrict__ w,
                                                 const float* __restrict__ bias,
                                                 float* __restrict__ out) {
  __shared__ bf16_t Al[2][64 * KT];
  __shared__ bf16_t Bl[2][64 * KT];
  int mt = blockIdx.x, nt = blockIdx.y;
  int t = threadIdx.x, wv = t >> 5, lane = t & 31;
  v8f acc[4];
#pragma unroll
  for (int j = 0; j < 4; ++j) acc[j] = zero8();

  const bf16_t* ga0 = x + (size_t)(mt * 64) * C_;
  const bf16_t* gb0 = w + (size_t)(nt * 64) * C_;
  const int niter = C_ / KT;

  if (wv == 0) {
    tdm_load_2d(lds_off32(&Al[0][0]), ga0, KT / 2, 64, C_ / 2, C_ / 2, 1u << 20);
    tdm_load_2d(lds_off32(&Bl[0][0]), gb0, KT / 2, 64, C_ / 2, C_ / 2, 1u << 20);
  }
  for (int it = 0; it < niter; ++it) {
    int cur = it & 1;
    __syncthreads();
    if (wv == 0) {
      if (it + 1 < niter) {
        tdm_load_2d(lds_off32(&Al[cur ^ 1][0]), ga0 + (it + 1) * KT, KT / 2, 64,
                    C_ / 2, C_ / 2, 1u << 20);
        tdm_load_2d(lds_off32(&Bl[cur ^ 1][0]), gb0 + (it + 1) * KT, KT / 2, 64,
                    C_ / 2, C_ / 2, 1u << 20);
        __builtin_amdgcn_s_wait_tensorcnt(2);
      } else {
        __builtin_amdgcn_s_wait_tensorcnt(0);
      }
    }
    __syncthreads();
    v16bf a = frag_rowk(&Al[cur][(wv * 16) * KT], KT);
#pragma unroll
    for (int j = 0; j < 4; ++j)
      acc[j] = wmma_bf(a, frag_rowk(&Bl[cur][(j * 16) * KT], KT), acc[j]);
  }

  int rbase = mt * 64 + wv * 16 + ((lane >> 4) << 3);
#pragma unroll
  for (int j = 0; j < 4; ++j) {
    int col = nt * 64 + j * 16 + (lane & 15);
    float bb = bias[col];
#pragma unroll
    for (int r = 0; r < 8; ++r)
      out[(size_t)(rbase + r) * C_ + col] = acc[j][r] + bb;
  }
}

// ---------- launcher ------------------------------------------------------

extern "C" void kernel_launch(void* const* d_in, const int* in_sizes, int n_in,
                              void* d_out, int out_size, void* d_ws, size_t ws_size,
                              hipStream_t stream) {
  (void)in_sizes; (void)n_in; (void)out_size; (void)ws_size;
  const float* x      = (const float*)d_in[0];
  const float* qkv_w  = (const float*)d_in[1];
  const float* proj_w = (const float*)d_in[2];
  const float* proj_b = (const float*)d_in[3];
  const float* dwc_w  = (const float*)d_in[4];
  const float* dwc_b  = (const float*)d_in[5];
  float* out = (float*)d_out;
  char* ws = (char*)d_ws;

  const size_t SXB = (size_t)B_ * N_ * C_ * 2;  // 102,760,448
  bf16_t* xb     = (bf16_t*)(ws);
  bf16_t* wqkvb  = (bf16_t*)(ws + SXB);
  bf16_t* wprojb = (bf16_t*)(ws + SXB + 1572864);
  bf16_t* qb     = (bf16_t*)(ws + SXB + 2097152);
  bf16_t* kb     = (bf16_t*)(ws + SXB + 2097152 + SXB);
  bf16_t* vb     = (bf16_t*)(ws + SXB + 2097152 + 2 * SXB);
  bf16_t* agentb = (bf16_t*)(ws + SXB + 2097152 + 3 * SXB);
  bf16_t* agentv = (bf16_t*)(ws + SXB + 2097152 + 3 * SXB + 2097152);
  bf16_t* dwcb   = kb;  // reuse: k dead after agent_attn
  bf16_t* outb   = xb;  // reuse: x dead after gemm_qkv

  int nx = B_ * N_ * C_;
  cvt_bf16<<<(nx + 255) / 256, 256, 0, stream>>>(x, xb, nx);
  cvt_bf16<<<(3 * C_ * C_ + 255) / 256, 256, 0, stream>>>(qkv_w, wqkvb, 3 * C_ * C_);
  cvt_bf16<<<(C_ * C_ + 255) / 256, 256, 0, stream>>>(proj_w, wprojb, C_ * C_);

  gemm_qkv<<<dim3((B_ * N_) / 64, (3 * C_) / 64), 128, 0, stream>>>(xb, wqkvb, qb, kb, vb);

  agent_pool<<<(B_ * AGP * C_) / 256, 256, 0, stream>>>(qb, agentb);

  agent_attn<<<B_ * HH, 128, 0, stream>>>(agentb, kb, vb, agentv);

  dwc_conv<<<(B_ * N_ * C_) / 256, 256, 0, stream>>>(vb, dwc_w, dwc_b, dwcb);

  q_attn<<<dim3(N_ / 64, HH, B_), 128, 0, stream>>>(qb, agentb, agentv, dwcb, outb);

  gemm_proj<<<dim3((B_ * N_) / 64, C_ / 64), 128, 0, stream>>>(outb, wprojb, proj_b, out);
}
